// MLA_65240553226463
// MI455X (gfx1250) — matrix operand order repaired and checked by reading, not compiled, not run on hardware
//
#include <hip/hip_runtime.h>
#include <hip/hip_bf16.h>

// ---------------- problem constants ----------------
#define B_DIM   4
#define S_DIM   2048
#define HID_DIM 2048
#define DOWN_DIM 512
#define UP_DIM  1024
#define ROPE_DIM 64
#define H_DIM   16
#define DK_DIM  128   // NOPE + ROPE
#define VD_DIM  64

typedef __attribute__((ext_vector_type(16))) __bf16   v16bf;
typedef __attribute__((ext_vector_type(8)))  float    v8f;
typedef __attribute__((ext_vector_type(4)))  unsigned u32x4;

union BF16Frag { v16bf v; u32x4 q[2]; };

__device__ __forceinline__ unsigned short f32_to_bf16_rne(float f) {
  unsigned u = __builtin_bit_cast(unsigned, f);
  u += 0x7FFFu + ((u >> 16) & 1u);           // round-to-nearest-even
  return (unsigned short)(u >> 16);
}
__device__ __forceinline__ unsigned pack_bf16x2(float lo, float hi) {
  return (unsigned)f32_to_bf16_rne(lo) | ((unsigned)f32_to_bf16_rne(hi) << 16);
}

// =====================================================================
// Generic GEMM: C[M,N] = A[M,K] @ B[K,N] + bias[N], fp32 in/out.
// Block tile 128x64 (BK=32), 8 waves (4M x 2N); each wave owns a 32x32
// macro-tile = 2x2 WMMA tiles -> 4 v_wmma per K-step, 2x fragment reuse.
// =====================================================================
#define GBM 128
#define GBN 64
#define GBK 32

__global__ __launch_bounds__(256) void gemm_bf16_wmma(
    const float* __restrict__ A, const float* __restrict__ Bm,
    const float* __restrict__ bias, float* __restrict__ C,
    int M, int N, int K) {
  __shared__ __align__(16) unsigned short As[GBM * GBK];  // [m][k]  8KB
  __shared__ __align__(16) unsigned short Bs[GBN * GBK];  // [n][k]  4KB
  const int tid = threadIdx.x;
  const int wid = tid >> 5, ln = tid & 31;
  const int lx = ln & 15, hf = ln >> 4, kg = hf << 3;
  const int bm = blockIdx.x * GBM, bn = blockIdx.y * GBN;
  const int wm = (wid & 3) << 5;   // 0,32,64,96
  const int wn = (wid >> 2) << 5;  // 0,32

  v8f acc[2][2] = {};
  for (int k0 = 0; k0 < K; k0 += GBK) {
    // A tile: 128x32 fp32, each thread converts 8 adjacent pairs -> u32 LDS stores
    #pragma unroll
    for (int i = 0; i < 8; ++i) {
      int e = tid + i * 256;                 // pair index, 2048 pairs
      int r = e >> 4, c = (e & 15) << 1;
      const float* ap = &A[(size_t)(bm + r) * K + (k0 + c)];
      *(unsigned*)&As[r * GBK + c] = pack_bf16x2(ap[0], ap[1]);
    }
    // B tile: 32x64 fp32 -> [n][k] transposed, k-pairs packed per u32
    #pragma unroll
    for (int i = 0; i < 4; ++i) {
      int e = tid + i * 256;                 // 1024 pairs
      int n = e & 63, kk = (e >> 6) << 1;
      float lo = Bm[(size_t)(k0 + kk) * N + (bn + n)];
      float hi = Bm[(size_t)(k0 + kk + 1) * N + (bn + n)];
      *(unsigned*)&Bs[n * GBK + kk] = pack_bf16x2(lo, hi);
    }
    __syncthreads();

    BF16Frag af[2], bfr[2];
    #pragma unroll
    for (int mi = 0; mi < 2; ++mi) {         // A: row per lane, K {kg..kg+7, 16+kg..}
      int ar = wm + mi * 16 + lx;
      af[mi].q[0] = *(const u32x4*)&As[ar * GBK + kg];
      af[mi].q[1] = *(const u32x4*)&As[ar * GBK + kg + 16];
    }
    #pragma unroll
    for (int ni = 0; ni < 2; ++ni) {         // B: col per lane, K {kg..kg+7, 16+kg..}
      int bc = wn + ni * 16 + lx;
      bfr[ni].q[0] = *(const u32x4*)&Bs[bc * GBK + kg];
      bfr[ni].q[1] = *(const u32x4*)&Bs[bc * GBK + kg + 16];
    }
    #pragma unroll
    for (int mi = 0; mi < 2; ++mi)
      #pragma unroll
      for (int ni = 0; ni < 2; ++ni)
        acc[mi][ni] = __builtin_amdgcn_wmma_f32_16x16x32_bf16(
            false, af[mi].v, false, bfr[ni].v, (short)0, acc[mi][ni], false, false);
    __syncthreads();
  }

  #pragma unroll
  for (int mi = 0; mi < 2; ++mi) {
    #pragma unroll
    for (int ni = 0; ni < 2; ++ni) {
      const int col = bn + wn + ni * 16 + lx;
      const float bv = bias[col];
      #pragma unroll
      for (int r = 0; r < 8; ++r) {
        int row = bm + wm + mi * 16 + (hf << 3) + r;
        C[(size_t)row * N + col] = acc[mi][ni][r] + bv;
      }
    }
  }
}

// =====================================================================
// RoPE + pack: build bf16 Q/K (B,H,S,128) and V (B,H,S,64) planes.
// Softmax scale folded into Q. One thread per (b,h,s); paired u32 stores.
// =====================================================================
__global__ __launch_bounds__(256) void pack_rope_qkv(
    const float* __restrict__ qc, const float* __restrict__ qrp,
    const float* __restrict__ kc, const float* __restrict__ krp,
    const float* __restrict__ vc,
    unsigned short* __restrict__ Qb, unsigned short* __restrict__ Kb,
    unsigned short* __restrict__ Vb) {
  const int idx = blockIdx.x * 256 + threadIdx.x;   // b*H*S
  const int s = idx & (S_DIM - 1);
  const int h = (idx >> 11) & (H_DIM - 1);
  const int b = idx >> 15;
  const float scale = 1.0f / (11.3137084989848f + 8.0f);  // 1/(sqrt(128)+sqrt(64))
  const size_t src   = (((size_t)(b * S_DIM + s)) * H_DIM + h) * 64;  // [b,s,h,64]
  const size_t dqk   = (((size_t)(b * H_DIM + h)) * S_DIM + s) * 128;
  const size_t dv    = (((size_t)(b * H_DIM + h)) * S_DIM + s) * 64;
  const size_t srcKr = ((size_t)(b * S_DIM + s)) * 64;

  #pragma unroll 4
  for (int d = 0; d < 64; d += 2) {
    *(unsigned*)&Qb[dqk + d] = pack_bf16x2(qc[src + d] * scale, qc[src + d + 1] * scale);
    *(unsigned*)&Kb[dqk + d] = pack_bf16x2(kc[src + d], kc[src + d + 1]);
    *(unsigned*)&Vb[dv + d]  = pack_bf16x2(vc[src + d], vc[src + d + 1]);
  }
  #pragma unroll 4
  for (int i = 0; i < 32; ++i) {
    const float inv = __expf(-0.287823136624256f * (float)i);  // 10000^(-2i/64)
    float sn, cs;
    __sincosf((float)s * inv, &sn, &cs);
    {
      float x1 = qrp[src + 2 * i], x2 = qrp[src + 2 * i + 1];
      *(unsigned*)&Qb[dqk + 64 + 2 * i] =
          pack_bf16x2((x1 * cs - x2 * sn) * scale, (x2 * cs + x1 * sn) * scale);
    }
    {
      float x1 = krp[srcKr + 2 * i], x2 = krp[srcKr + 2 * i + 1];
      *(unsigned*)&Kb[dqk + 64 + 2 * i] =
          pack_bf16x2(x1 * cs - x2 * sn, x2 * cs + x1 * sn);
    }
  }
}

// =====================================================================
// Flash attention: block = 128 q-rows of one (b,h); wave = 16 rows.
// Q resident in VGPRs; K tile staged via GLOBAL_LOAD_ASYNC_TO_LDS_B128
// (ASYNCcnt path, no VGPR round-trip); V transposed manually into LDS.
// Online softmax with shfl_xor reductions inside 16-lane halves.
// =====================================================================
__global__ __launch_bounds__(256) void mla_attention(
    const unsigned short* __restrict__ Qb, const unsigned short* __restrict__ Kb,
    const unsigned short* __restrict__ Vb, float* __restrict__ Out) {
  __shared__ __align__(16) unsigned short Ks[64 * 128];    // [kpos][d]      16KB
  __shared__ __align__(16) unsigned short Vs[64 * 64];     // [vd][kpos]      8KB
  __shared__ __align__(16) unsigned short Ps[8][16 * 64];  // per-wave P pad 16KB
  const int tid = threadIdx.x, wid = tid >> 5, ln = tid & 31;
  const int lx = ln & 15, hf = ln >> 4, kg = hf << 3;
  const int q0 = blockIdx.x * 128;
  const int bh = blockIdx.y;
  const int h = bh & (H_DIM - 1), b = bh >> 4;
  const size_t qkBase = (size_t)bh * S_DIM * 128;
  const size_t vBase  = (size_t)bh * S_DIM * 64;

  // Q fragments resident in registers (A-matrix layout, straight from global)
  BF16Frag qf[4];
  {
    const int qrow = q0 + wid * 16 + lx;
    const u32x4* qp = (const u32x4*)&Qb[qkBase + (size_t)qrow * 128];
    #pragma unroll
    for (int dc = 0; dc < 4; ++dc) {
      qf[dc].q[0] = qp[dc * 4 + hf];
      qf[dc].q[1] = qp[dc * 4 + hf + 2];
    }
  }

  v8f o[4] = {};
  float m_run[8], l_run[8];
  #pragma unroll
  for (int r = 0; r < 8; ++r) { m_run[r] = -1e30f; l_run[r] = 0.0f; }

  const int jmax = (q0 + 127) >> 6;
  for (int j = 0; j <= jmax; ++j) {
    const int kbase = j << 6;
    {
      // --- K tile: async memory->LDS DMA, no VGPR staging (ASYNCcnt) ---
      const unsigned short* ksrc = &Kb[qkBase + (size_t)kbase * 128];
      #pragma unroll
      for (int i = 0; i < 4; ++i) {
        int e = tid + i * 256;                        // 1024 x 16B = 16KB
        unsigned ldsa = (unsigned)(size_t)&Ks[e * 8]; // low 32b of flat = LDS addr
        unsigned voff = (unsigned)(e * 16);
        asm volatile("global_load_async_to_lds_b128 %0, %1, %2"
                     :: "v"(ldsa), "v"(voff), "s"(ksrc) : "memory");
      }
      // --- V tile: manual copy with transpose into [vd][kpos] ---
      const u32x4* vgp = (const u32x4*)&Vb[vBase + (size_t)kbase * 64];
      #pragma unroll
      for (int i = 0; i < 2; ++i) {
        int e = tid + i * 256;
        u32x4 t = vgp[e];
        int kpos = e >> 3;
        int vd0 = (e & 7) << 3;
        const unsigned short* ts = (const unsigned short*)&t;
        #pragma unroll
        for (int x = 0; x < 8; ++x) Vs[(vd0 + x) * 64 + kpos] = ts[x];
      }
      asm volatile("s_wait_asynccnt 0" ::: "memory");
    }
    __syncthreads();

    // S = Q @ K^T  (16 x 64 per wave, d accumulated over 128)
    v8f sc[4];
    #pragma unroll
    for (int nt = 0; nt < 4; ++nt) {
      v8f a = {};
      const int col = nt * 16 + lx;
      #pragma unroll
      for (int dc = 0; dc < 4; ++dc) {
        BF16Frag kf;
        kf.q[0] = *(const u32x4*)&Ks[col * 128 + dc * 32 + kg];
        kf.q[1] = *(const u32x4*)&Ks[col * 128 + dc * 32 + kg + 16];
        a = __builtin_amdgcn_wmma_f32_16x16x32_bf16(false, qf[dc].v, false, kf.v,
                                                    (short)0, a, false, false);
      }
      sc[nt] = a;
    }

    // causal mask + row max
    float mt[8];
    #pragma unroll
    for (int r = 0; r < 8; ++r) {
      const int rowg = q0 + wid * 16 + (hf << 3) + r;
      float mx = -1e30f;
      #pragma unroll
      for (int nt = 0; nt < 4; ++nt) {
        const int colg = kbase + nt * 16 + lx;
        float v = sc[nt][r];
        if (colg > rowg) v = -1e9f;
        sc[nt][r] = v;
        mx = fmaxf(mx, v);
      }
      mt[r] = mx;
    }
    #pragma unroll
    for (int msk = 1; msk <= 8; msk <<= 1) {   // stays inside 16-lane halves
      #pragma unroll
      for (int r = 0; r < 8; ++r)
        mt[r] = fmaxf(mt[r], __shfl_xor(mt[r], msk, 32));
    }

    float alpha[8], ls[8];
    #pragma unroll
    for (int r = 0; r < 8; ++r) {
      float mnew = fmaxf(m_run[r], mt[r]);
      alpha[r] = __expf(m_run[r] - mnew);
      m_run[r] = mnew;
      ls[r] = 0.0f;
    }
    #pragma unroll
    for (int nt = 0; nt < 4; ++nt) {
      #pragma unroll
      for (int r = 0; r < 8; ++r) {
        float p = __expf(sc[nt][r] - m_run[r]);
        ls[r] += p;
        Ps[wid][((hf << 3) + r) * 64 + nt * 16 + lx] = f32_to_bf16_rne(p);
      }
    }
    #pragma unroll
    for (int msk = 1; msk <= 8; msk <<= 1) {
      #pragma unroll
      for (int r = 0; r < 8; ++r)
        ls[r] += __shfl_xor(ls[r], msk, 32);
    }
    #pragma unroll
    for (int r = 0; r < 8; ++r) l_run[r] = l_run[r] * alpha[r] + ls[r];
    #pragma unroll
    for (int vt = 0; vt < 4; ++vt) {
      #pragma unroll
      for (int r = 0; r < 8; ++r) o[vt][r] *= alpha[r];
    }

    // O += P @ V   (K=64 -> two K=32 WMMAs per vd tile); Ps RAW is
    // wave-private: DS ops from one wave are in-order, no barrier needed.
    #pragma unroll
    for (int vt = 0; vt < 4; ++vt) {
      const int vcol = vt * 16 + lx;
      #pragma unroll
      for (int kc = 0; kc < 2; ++kc) {
        BF16Frag pf, vf;
        pf.q[0] = *(const u32x4*)&Ps[wid][lx * 64 + kc * 32 + kg];
        pf.q[1] = *(const u32x4*)&Ps[wid][lx * 64 + kc * 32 + kg + 16];
        vf.q[0] = *(const u32x4*)&Vs[vcol * 64 + kc * 32 + kg];
        vf.q[1] = *(const u32x4*)&Vs[vcol * 64 + kc * 32 + kg + 16];
        o[vt] = __builtin_amdgcn_wmma_f32_16x16x32_bf16(false, pf.v, false, vf.v,
                                                        (short)0, o[vt], false, false);
      }
    }
    __syncthreads();
  }

  float linv[8];
  #pragma unroll
  for (int r = 0; r < 8; ++r) linv[r] = 1.0f / l_run[r];
  #pragma unroll
  for (int vt = 0; vt < 4; ++vt) {
    const int colo = h * 64 + vt * 16 + lx;
    #pragma unroll
    for (int r = 0; r < 8; ++r) {
      const int rowg = q0 + wid * 16 + (hf << 3) + r;
      Out[((size_t)(b * S_DIM + rowg)) * UP_DIM + colo] = o[vt][r] * linv[r];
    }
  }
}

// =====================================================================
extern "C" void kernel_launch(void* const* d_in, const int* in_sizes, int n_in,
                              void* d_out, int out_size, void* d_ws, size_t ws_size,
                              hipStream_t stream) {
  (void)in_sizes; (void)n_in; (void)out_size; (void)ws_size;
  const float* X     = (const float*)d_in[0];
  // d_in[1] = mask (causal tril) — handled analytically in the kernel
  const float* W_dkv = (const float*)d_in[2];
  const float* b_dkv = (const float*)d_in[3];
  const float* W_dq  = (const float*)d_in[4];
  const float* b_dq  = (const float*)d_in[5];
  const float* W_uk  = (const float*)d_in[6];
  const float* b_uk  = (const float*)d_in[7];
  const float* W_uv  = (const float*)d_in[8];
  const float* b_uv  = (const float*)d_in[9];
  const float* W_uq  = (const float*)d_in[10];
  const float* b_uq  = (const float*)d_in[11];
  const float* W_qr  = (const float*)d_in[12];
  const float* b_qr  = (const float*)d_in[13];
  const float* W_kr  = (const float*)d_in[14];
  const float* b_kr  = (const float*)d_in[15];
  const float* W_fc  = (const float*)d_in[16];
  const float* b_fc  = (const float*)d_in[17];
  float* out = (float*)d_out;

  const int BS = B_DIM * S_DIM;  // 8192
  char* ws = (char*)d_ws;
  float* c_kv = (float*)ws;  ws += (size_t)BS * DOWN_DIM * 4;
  float* c_q  = (float*)ws;  ws += (size_t)BS * DOWN_DIM * 4;
  float* k_c  = (float*)ws;  ws += (size_t)BS * UP_DIM * 4;
  float* v_c  = (float*)ws;  ws += (size_t)BS * UP_DIM * 4;
  float* q_c  = (float*)ws;  ws += (size_t)BS * UP_DIM * 4;
  float* q_r  = (float*)ws;  ws += (size_t)BS * UP_DIM * 4;
  float* k_r  = (float*)ws;  ws += (size_t)BS * ROPE_DIM * 4;
  unsigned short* Qb = (unsigned short*)ws; ws += (size_t)BS * H_DIM * DK_DIM * 2;
  unsigned short* Kb = (unsigned short*)ws; ws += (size_t)BS * H_DIM * DK_DIM * 2;
  unsigned short* Vb = (unsigned short*)ws; ws += (size_t)BS * H_DIM * VD_DIM * 2;
  float* attn_out = k_c;  // k_c fully consumed by pack_rope_qkv before attention

  dim3 blk(256);
  // low-rank down projections
  gemm_bf16_wmma<<<dim3(BS / GBM, DOWN_DIM / GBN), blk, 0, stream>>>(
      X, W_dkv, b_dkv, c_kv, BS, DOWN_DIM, HID_DIM);
  gemm_bf16_wmma<<<dim3(BS / GBM, DOWN_DIM / GBN), blk, 0, stream>>>(
      X, W_dq, b_dq, c_q, BS, DOWN_DIM, HID_DIM);
  // up projections
  gemm_bf16_wmma<<<dim3(BS / GBM, UP_DIM / GBN), blk, 0, stream>>>(
      c_kv, W_uk, b_uk, k_c, BS, UP_DIM, DOWN_DIM);
  gemm_bf16_wmma<<<dim3(BS / GBM, UP_DIM / GBN), blk, 0, stream>>>(
      c_kv, W_uv, b_uv, v_c, BS, UP_DIM, DOWN_DIM);
  gemm_bf16_wmma<<<dim3(BS / GBM, UP_DIM / GBN), blk, 0, stream>>>(
      c_q, W_uq, b_uq, q_c, BS, UP_DIM, DOWN_DIM);
  gemm_bf16_wmma<<<dim3(BS / GBM, UP_DIM / GBN), blk, 0, stream>>>(
      c_q, W_qr, b_qr, q_r, BS, UP_DIM, DOWN_DIM);
  gemm_bf16_wmma<<<dim3(BS / GBM, ROPE_DIM / GBN), blk, 0, stream>>>(
      X, W_kr, b_kr, k_r, BS, ROPE_DIM, HID_DIM);
  // rope + bf16 packing into (B,H,S,d) planes
  pack_rope_qkv<<<dim3((B_DIM * H_DIM * S_DIM) / 256), blk, 0, stream>>>(
      q_c, q_r, k_c, k_r, v_c, Qb, Kb, Vb);
  // fused causal flash attention
  mla_attention<<<dim3(S_DIM / 128, B_DIM * H_DIM), blk, 0, stream>>>(
      Qb, Kb, Vb, attn_out);
  // output projection
  gemm_bf16_wmma<<<dim3(BS / GBM, HID_DIM / GBN), blk, 0, stream>>>(
      attn_out, W_fc, b_fc, out, BS, HID_DIM, UP_DIM);
}
